// DecoderOnlyExpander_8821862825992
// MI455X (gfx1250) — matrix operand order repaired and checked by reading, not compiled, not run on hardware
//
#include <hip/hip_runtime.h>
#include <math.h>

// ---------------------------------------------------------------------------
// CDNA5 (gfx1250) wave32 WMMA transformer forward.
// - All matmuls (projections, FFN, QK^T, PV, output head) run on
//   v_wmma_f32_16x16x32_f16 (f16 inputs, f32 accumulate).
// - Straight-copy tiles (GEMM A, attention Q/K) use the CDNA5 async
//   global->LDS DMA path (global_load_async_to_lds_b128 + s_wait_asynccnt).
// ---------------------------------------------------------------------------

typedef _Float16 h16;
typedef __attribute__((ext_vector_type(16))) _Float16 v16h;
typedef __attribute__((ext_vector_type(8)))  float    v8f;

// one async b128 global->LDS copy; imm offset applies to BOTH addresses
#define ASYNC_B128(ldsoff, gptr, imm)                                          \
  asm volatile("global_load_async_to_lds_b128 %0, %1, off offset:" #imm        \
               :: "v"(ldsoff), "v"(gptr) : "memory")

__device__ inline void wait_async0() {
#if defined(__has_builtin)
#if __has_builtin(__builtin_amdgcn_s_wait_asynccnt)
  __builtin_amdgcn_s_wait_asynccnt(0);
#else
  asm volatile("s_wait_asynccnt 0" ::: "memory");
#endif
#else
  asm volatile("s_wait_asynccnt 0" ::: "memory");
#endif
}

__device__ inline unsigned lds_off32(const void* p) {
  // generic LDS pointer: addr[31:0] is the LDS byte offset (ISA 10.2)
  return (unsigned)(unsigned long long)p;
}

__device__ inline v8f zero8() {
  v8f z;
#pragma unroll
  for (int i = 0; i < 8; i++) z[i] = 0.f;
  return z;
}

__device__ inline v8f wmma16(v16h a, v16h b, v8f c) {
  // D = A(16x32 f16) * B(32x16 f16) + C(16x16 f32)
  return __builtin_amdgcn_wmma_f32_16x16x32_f16(
      /*neg_a=*/false, a, /*neg_b=*/false, b,
      /*c_mod=*/(short)0, c, /*reuse_a=*/false, /*reuse_b=*/false);
}

// A-fragment (16x32, MxK) from LDS tile stored row-major [M][ldk].
// Lanes 0-15: row=lane, K=k0..k0+15;
// lanes 16-31: row=lane-16, K=k0+16..k0+31 (matches lds_load_b).
__device__ inline v16h lds_load_a(const h16* base, int ldk, int m0, int k0, int lane) {
  const h16* p = base + (size_t)(m0 + (lane & 15)) * ldk + k0 + ((lane >> 4) << 4);
  v16h a;
#pragma unroll
  for (int i = 0; i < 8; i++) a[i] = p[i];
#pragma unroll
  for (int i = 0; i < 8; i++) a[8 + i] = p[8 + i];
  return a;
}

// B-fragment (32x16, KxN) from LDS tile stored N-major [N][ldk] (i.e. B^T).
// lanes 0-15: col=n0+lane, K=k0..k0+15 ; lanes 16-31: col, K=k0+16..k0+31.
__device__ inline v16h lds_load_b(const h16* base, int ldk, int n0, int k0, int lane) {
  const h16* p = base + (size_t)(n0 + (lane & 15)) * ldk + k0 + ((lane >> 4) << 4);
  v16h b;
#pragma unroll
  for (int i = 0; i < 16; i++) b[i] = p[i];
  return b;
}

// ---------------------------------------------------------------------------
// Generic f16 GEMM: C[MxN] = A[MxK](f16,row) * B[KxN](f16,row) (+bias)(+res)
// Block: 256 thr = 8 waves (2x4), tile 64x128, K-step 64. Each wave: 32x32.
// A tile copied by async DMA; B tile transposed through VGPRs (overlapped).
// ---------------------------------------------------------------------------
__global__ __launch_bounds__(256) void gemm_f16_kernel(
    const h16* __restrict__ A, const h16* __restrict__ Bm,
    const float* __restrict__ bias, const float* __restrict__ res,
    float* __restrict__ Cf, h16* __restrict__ Ch,
    int M, int N, int K)
{
  __shared__ h16 sA[64][64];    // 8 KB
  __shared__ h16 sBT[128][64];  // 16 KB, B transposed: [n][k]
  const int tid = threadIdx.x;
  const int lane = tid & 31;
  const int wave = tid >> 5;
  const int wm = wave >> 2;     // 0..1
  const int wn = wave & 3;      // 0..3
  const int bm = blockIdx.y * 64;
  const int bn = blockIdx.x * 128;

  v8f acc[2][2];
#pragma unroll
  for (int mi = 0; mi < 2; mi++)
#pragma unroll
    for (int ni = 0; ni < 2; ni++) acc[mi][ni] = zero8();

  const bool fullN = ((bn + 128) <= N) && ((N & 7) == 0);

  for (int k0 = 0; k0 < K; k0 += 64) {
    // ---- A tile 64x64: async global->LDS DMA (32 B per thread) ----
    {
      const int r = tid >> 2;
      const int cg = (tid & 3) << 4;
      const h16* gp = A + (size_t)(bm + r) * K + k0 + cg;
      const unsigned lo = lds_off32(&sA[r][cg]);
      ASYNC_B128(lo, gp, 0);
      ASYNC_B128(lo, gp, 16);
    }
    // ---- B tile 64x128, transpose-store (overlaps with A DMA) ----
    {
      const int r = tid >> 4;
      const int cg = (tid & 15) << 3;
#pragma unroll
      for (int rr = 0; rr < 4; rr++) {
        const int row = r + rr * 16;
        if (fullN) {
          union { uint4 u; h16 h[8]; } t;
          t.u = *(const uint4*)(Bm + (size_t)(k0 + row) * N + bn + cg);
#pragma unroll
          for (int e = 0; e < 8; e++) sBT[cg + e][row] = t.h[e];
        } else {
          // clamped loads + select: no exec-mask churn, always in-bounds
#pragma unroll
          for (int e = 0; e < 8; e++) {
            const int col = bn + cg + e;
            const int colc = (col < N) ? col : (N - 1);
            const h16 v = Bm[(size_t)(k0 + row) * N + colc];
            sBT[cg + e][row] = (col < N) ? v : (h16)0.f;
          }
        }
      }
    }
    if (k0 + 64 < K) {  // hint next tiles into cache (global_prefetch_b8)
      __builtin_prefetch(A + (size_t)(bm + (tid >> 2)) * K + k0 + 64, 0, 1);
      __builtin_prefetch(Bm + (size_t)(k0 + 64 + (tid >> 4)) * N + bn, 0, 1);
    }
    wait_async0();
    __syncthreads();

    const int m0 = wm * 32;
    const int n0 = wn * 32;
#pragma unroll
    for (int kk = 0; kk < 64; kk += 32) {
      v16h a0 = lds_load_a(&sA[0][0], 64, m0, kk, lane);
      v16h a1 = lds_load_a(&sA[0][0], 64, m0 + 16, kk, lane);
      v16h b0 = lds_load_b(&sBT[0][0], 64, n0, kk, lane);
      v16h b1 = lds_load_b(&sBT[0][0], 64, n0 + 16, kk, lane);
      acc[0][0] = wmma16(a0, b0, acc[0][0]);
      acc[0][1] = wmma16(a0, b1, acc[0][1]);
      acc[1][0] = wmma16(a1, b0, acc[1][0]);
      acc[1][1] = wmma16(a1, b1, acc[1][1]);
    }
    __syncthreads();
  }

  // ---- epilogue: C/D layout: VGPR i -> row i (lanes<16) / row 8+i (lanes>=16)
#pragma unroll
  for (int mi = 0; mi < 2; mi++)
#pragma unroll
    for (int ni = 0; ni < 2; ni++)
#pragma unroll
      for (int i = 0; i < 8; i++) {
        const int row = bm + wm * 32 + mi * 16 + ((lane >> 4) << 3) + i;
        const int col = bn + wn * 32 + ni * 16 + (lane & 15);
        if (col < N) {
          float v = acc[mi][ni][i];
          if (bias) v += bias[col];
          if (res) v += res[(size_t)row * N + col];
          if (Cf) Cf[(size_t)row * N + col] = v;
          if (Ch) Ch[(size_t)row * N + col] = (h16)v;
        }
      }
}

// ---------------------------------------------------------------------------
// Flash-style attention (self causal when seg==nullptr, else seg-window).
// Block: 128 thr = 4 waves; 64 queries/block (16 per wave), d_head = 64.
// QK^T and P*V both on WMMA f16. Online softmax in f32.
// Q and K tiles use async global->LDS DMA; V transposed through VGPRs.
// ---------------------------------------------------------------------------
__global__ __launch_bounds__(128) void attn_kernel(
    const h16* __restrict__ Q, const h16* __restrict__ Km,
    const h16* __restrict__ Vm, h16* __restrict__ O,
    const int* __restrict__ seg, int H, int Lq, int Lk, int D,
    float scale, int lookback)
{
  __shared__ h16 sQ[64][64];      // [q][d]   8 KB
  __shared__ h16 sK[64][64];      // [key][d] 8 KB  (== B^T for scores)
  __shared__ h16 sVT[64][64];     // [d][key] 8 KB  (== B^T for PV)
  __shared__ h16 sP[4][16][64];   // per-wave probs 8 KB

  const int tid = threadIdx.x;
  const int lane = tid & 31;
  const int wave = tid >> 5;
  const int qblk = blockIdx.x;
  const int bh = blockIdx.y;
  const int b = bh / H;
  const int hh = bh % H;
  const int qbase = qblk * 64;

  // ---- load Q tile: async DMA, 64 B per thread ----
  {
    const int r = tid >> 1;
    const int hf = tid & 1;
    const h16* gq = Q + (size_t)(b * Lq + qbase + r) * D + hh * 64 + hf * 32;
    const unsigned lq = lds_off32(&sQ[r][hf * 32]);
    ASYNC_B128(lq, gq, 0);
    ASYNC_B128(lq, gq, 16);
    ASYNC_B128(lq, gq, 32);
    ASYNC_B128(lq, gq, 48);
  }
  wait_async0();
  __syncthreads();

  v16h qa0 = lds_load_a(&sQ[0][0], 64, wave * 16, 0, lane);
  v16h qa1 = lds_load_a(&sQ[0][0], 64, wave * 16, 32, lane);

  float rowmax[8], rowsum[8];
  v8f o[4];
#pragma unroll
  for (int i = 0; i < 8; i++) { rowmax[i] = -1e30f; rowsum[i] = 0.f; }
#pragma unroll
  for (int n = 0; n < 4; n++) o[n] = zero8();

  const int qrow_lo = qbase + wave * 16 + ((lane >> 4) << 3);
  int segv[8];
  if (seg) {
#pragma unroll
    for (int i = 0; i < 8; i++) segv[i] = seg[b * Lq + qrow_lo + i];
  }
  const int nkb = seg ? (Lk >> 6) : (qblk + 1);

  for (int kb = 0; kb < nkb; kb++) {
    __syncthreads();
    // ---- K tile: async DMA; V tile: manual transpose through VGPRs ----
    {
      const int r = tid >> 1;
      const int hf = tid & 1;
      const size_t krow = (size_t)(b * Lk + kb * 64 + r) * D + hh * 64 + hf * 32;
      const h16* gk = Km + krow;
      const unsigned lk = lds_off32(&sK[r][hf * 32]);
      ASYNC_B128(lk, gk, 0);
      ASYNC_B128(lk, gk, 16);
      ASYNC_B128(lk, gk, 32);
      ASYNC_B128(lk, gk, 48);
      union { uint4 u[4]; h16 h[32]; } tv;
      const uint4* vs = (const uint4*)(Vm + krow);
#pragma unroll
      for (int i = 0; i < 4; i++) tv.u[i] = vs[i];
#pragma unroll
      for (int e = 0; e < 32; e++) sVT[hf * 32 + e][r] = tv.h[e];
    }
    wait_async0();
    __syncthreads();

    // ---- scores S = Q * K^T  (16 x 64 per wave) ----
    v8f s[4];
#pragma unroll
    for (int j = 0; j < 4; j++) s[j] = zero8();
#pragma unroll
    for (int j = 0; j < 4; j++) {
      v16h bk0 = lds_load_b(&sK[0][0], 64, j * 16, 0, lane);
      v16h bk1 = lds_load_b(&sK[0][0], 64, j * 16, 32, lane);
      s[j] = wmma16(qa0, bk0, s[j]);
      s[j] = wmma16(qa1, bk1, s[j]);
    }

    // ---- mask + online softmax update ----
    const int klane = kb * 64 + (lane & 15);
#pragma unroll
    for (int i = 0; i < 8; i++) {
      const int qrow = qrow_lo + i;
      float m = -1e30f;
#pragma unroll
      for (int j = 0; j < 4; j++) {
        const int kidx = klane + j * 16;
        float val = s[j][i] * scale;
        const bool ok = seg ? (kidx <= segv[i] && kidx > segv[i] - lookback)
                            : (kidx <= qrow);
        val = ok ? val : -1e30f;
        s[j][i] = val;
        m = fmaxf(m, val);
      }
#pragma unroll
      for (int off = 8; off >= 1; off >>= 1) m = fmaxf(m, __shfl_xor(m, off, 32));
      const float nm = fmaxf(rowmax[i], m);
      const float alpha = __expf(rowmax[i] - nm);
      float psum = 0.f;
#pragma unroll
      for (int j = 0; j < 4; j++) {
        const float sv = s[j][i];
        const float p = (sv <= -1e29f) ? 0.f : __expf(sv - nm);
        sP[wave][i + ((lane >> 4) << 3)][j * 16 + (lane & 15)] = (h16)p;
        psum += p;
      }
#pragma unroll
      for (int off = 8; off >= 1; off >>= 1) psum += __shfl_xor(psum, off, 32);
      rowsum[i] = rowsum[i] * alpha + psum;
      rowmax[i] = nm;
#pragma unroll
      for (int n = 0; n < 4; n++) o[n][i] *= alpha;
    }
    __syncthreads();

    // ---- O += P * V ----
    v16h pa0 = lds_load_a(&sP[wave][0][0], 64, 0, 0, lane);
    v16h pa1 = lds_load_a(&sP[wave][0][0], 64, 0, 32, lane);
#pragma unroll
    for (int n = 0; n < 4; n++) {
      v16h bv0 = lds_load_b(&sVT[0][0], 64, n * 16, 0, lane);
      v16h bv1 = lds_load_b(&sVT[0][0], 64, n * 16, 32, lane);
      o[n] = wmma16(pa0, bv0, o[n]);
      o[n] = wmma16(pa1, bv1, o[n]);
    }
  }

  // ---- normalize + store f16 ----
#pragma unroll
  for (int n = 0; n < 4; n++)
#pragma unroll
    for (int i = 0; i < 8; i++) {
      const int qrow = qrow_lo + i;
      const float l = rowsum[i];
      const float val = (l > 0.f) ? o[n][i] / l : 0.f;
      O[(size_t)(b * Lq + qrow) * D + hh * 64 + n * 16 + (lane & 15)] = (h16)val;
    }
}

// ---------------------------------------------------------------------------
// Elementwise / small kernels
// ---------------------------------------------------------------------------
__global__ __launch_bounds__(128) void rmsnorm_kernel(
    const float* __restrict__ X, const float* __restrict__ W,
    h16* __restrict__ Y, int D)
{
  const int row = blockIdx.x;
  const int tid = threadIdx.x;
  const float* x = X + (size_t)row * D;
  float ss = 0.f;
  for (int i = tid; i < D; i += 128) { const float v = x[i]; ss += v * v; }
#pragma unroll
  for (int off = 16; off >= 1; off >>= 1) ss += __shfl_xor(ss, off, 32);
  __shared__ float red[4];
  if ((tid & 31) == 0) red[tid >> 5] = ss;
  __syncthreads();
  const float tot = red[0] + red[1] + red[2] + red[3];
  const float r = rsqrtf(tot / (float)D + 1e-6f);
  for (int i = tid; i < D; i += 128) Y[(size_t)row * D + i] = (h16)(x[i] * r * W[i]);
}

__global__ void rope_kernel(const float* __restrict__ Qf, const float* __restrict__ Kf,
                            h16* __restrict__ Qh, h16* __restrict__ Kh,
                            int total_tok, int L, int H, int dh)
{
  const int d2 = dh >> 1;
  const int idx = blockIdx.x * blockDim.x + threadIdx.x;
  if (idx >= total_tok * H * d2) return;
  const int i = idx % d2;
  const int hh = (idx / d2) % H;
  const int tok = idx / (d2 * H);
  const int l = tok % L;
  const float inv = __expf(-(float)i * (logf(10000.f) * 2.f / (float)dh));
  float sn, cs;
  __sincosf((float)l * inv, &sn, &cs);
  const size_t base = (size_t)tok * (H * dh) + hh * dh;
  const float q1 = Qf[base + i], q2 = Qf[base + i + d2];
  Qh[base + i]      = (h16)(q1 * cs - q2 * sn);
  Qh[base + i + d2] = (h16)(q2 * cs + q1 * sn);
  const float k1 = Kf[base + i], k2 = Kf[base + i + d2];
  Kh[base + i]      = (h16)(k1 * cs - k2 * sn);
  Kh[base + i + d2] = (h16)(k2 * cs + k1 * sn);
}

__global__ void f2h_kernel(const float* __restrict__ src, h16* __restrict__ dst, size_t n)
{
  const size_t idx = (size_t)blockIdx.x * blockDim.x + threadIdx.x;
  if (idx < n) dst[idx] = (h16)src[idx];
}

__global__ void swiglu_kernel(const float* __restrict__ h1, const float* __restrict__ h3,
                              h16* __restrict__ out, size_t n)
{
  const size_t idx = (size_t)blockIdx.x * blockDim.x + threadIdx.x;
  if (idx < n) {
    const float a = h1[idx];
    const float g = a / (1.f + __expf(-a));
    out[idx] = (h16)(g * h3[idx]);
  }
}

__global__ void embed_lo_kernel(const float* __restrict__ emb, const int* __restrict__ codes,
                                float* __restrict__ X, int B, int L, int D)
{
  const size_t idx = (size_t)blockIdx.x * blockDim.x + threadIdx.x;
  if (idx >= (size_t)B * L * D) return;
  const int dd = (int)(idx % D);
  const size_t tok = idx / D;
  const int b = (int)(tok / L);
  const int l = (int)(tok % L);
  const int id = (l == 0) ? 257 : codes[b * L + l - 1];  // EOS shift
  X[idx] = emb[(size_t)id * D + dd];
}

__global__ void embed_hi_kernel(const float* __restrict__ emb, const int* __restrict__ codes,
                                h16* __restrict__ Mb, int n_tok, int D)
{
  const size_t idx = (size_t)blockIdx.x * blockDim.x + threadIdx.x;
  if (idx >= (size_t)n_tok * D) return;
  const int dd = (int)(idx % D);
  const int tok = (int)(idx / D);
  Mb[idx] = (h16)emb[(size_t)codes[tok] * D + dd];
}

// ---------------------------------------------------------------------------
// Host driver
// ---------------------------------------------------------------------------
extern "C" void kernel_launch(void* const* d_in, const int* in_sizes, int n_in,
                              void* d_out, int out_size, void* d_ws, size_t ws_size,
                              hipStream_t stream)
{
  (void)in_sizes; (void)n_in; (void)out_size; (void)ws_size;
  const int B = 4, L = 2048, LH = 256, D = 512, DFF = 2048, H = 8, DH = 64, NL = 6, KLO = 260;
  const int M = B * L;      // 8192 decoder tokens
  const int Mh = B * LH;    // 1024 memory tokens
  const float scale = 0.125f;  // 1/sqrt(64)

  const int* codes_hi = (const int*)d_in[0];
  const int* codes_lo = (const int*)d_in[1];
  const int* seg_ids  = (const int*)d_in[2];
  const float* emb_hi = (const float*)d_in[3];
  const float* emb_lo = (const float*)d_in[4];
  // layer l param j: insertion order n1,n2,n3, sa{wq,bq,wk,bk,wv,bv,wo,bo}, ca{wq,wk,wv,wo}, ffn{w1,w3,w2}
  auto LP = [&](int l, int j) -> const float* { return (const float*)d_in[5 + l * 18 + j]; };
  const float* fn_w  = (const float*)d_in[5 + NL * 18 + 0];
  const float* out_w = (const float*)d_in[5 + NL * 18 + 1];
  const float* out_b = (const float*)d_in[5 + NL * 18 + 2];

  // bump allocator over workspace
  size_t off = 0;
  auto alloc = [&](size_t bytes) -> void* {
    off = (off + 255) & ~(size_t)255;
    void* p = (char*)d_ws + off;
    off += bytes;
    return p;
  };

  h16*   memb = (h16*)alloc((size_t)Mh * D * 2);
  float* x    = (float*)alloc((size_t)M * D * 4);
  h16*   xn   = (h16*)alloc((size_t)M * D * 2);
  float* qf   = (float*)alloc((size_t)M * D * 4);
  float* kf   = (float*)alloc((size_t)M * D * 4);
  h16*   qb   = (h16*)alloc((size_t)M * D * 2);
  h16*   kb   = (h16*)alloc((size_t)M * D * 2);
  h16*   vb   = (h16*)alloc((size_t)M * D * 2);
  h16*   ob   = (h16*)alloc((size_t)M * D * 2);
  h16*   kcb  = (h16*)alloc((size_t)Mh * D * 2);
  h16*   vcb  = (h16*)alloc((size_t)Mh * D * 2);
  float* h1   = (float*)alloc((size_t)M * DFF * 4);
  float* h3   = (float*)alloc((size_t)M * DFF * 4);
  h16*   hf   = (h16*)alloc((size_t)M * DFF * 2);

  // f16 weight copies
  const int    widx[11] = {3, 5, 7, 9, 11, 12, 13, 14, 15, 16, 17};
  const size_t DD = (size_t)D * D;
  const size_t wsz[11] = {DD, DD, DD, DD, DD, DD, DD, DD,
                          (size_t)D * DFF, (size_t)D * DFF, (size_t)DFF * D};
  h16* wb[6][11];
  for (int l = 0; l < NL; l++)
    for (int j = 0; j < 11; j++) {
      wb[l][j] = (h16*)alloc(wsz[j] * 2);
      f2h_kernel<<<(unsigned)((wsz[j] + 255) / 256), 256, 0, stream>>>(LP(l, widx[j]), wb[l][j], wsz[j]);
    }
  h16* outw = (h16*)alloc((size_t)D * KLO * 2);
  f2h_kernel<<<(unsigned)(((size_t)D * KLO + 255) / 256), 256, 0, stream>>>(out_w, outw, (size_t)D * KLO);

  // embeddings
  embed_lo_kernel<<<(unsigned)(((size_t)M * D + 255) / 256), 256, 0, stream>>>(emb_lo, codes_lo, x, B, L, D);
  embed_hi_kernel<<<(unsigned)(((size_t)Mh * D + 255) / 256), 256, 0, stream>>>(emb_hi, codes_hi, memb, Mh, D);

  auto gemm = [&](const h16* A, const h16* Bm, const float* bias, const float* res,
                  float* Cf, h16* Ch, int Mm, int Nn, int Kk) {
    dim3 grid((Nn + 127) / 128, Mm / 64);
    gemm_f16_kernel<<<grid, 256, 0, stream>>>(A, Bm, bias, res, Cf, Ch, Mm, Nn, Kk);
  };

  const int npairs = M * H * (DH / 2);
  for (int l = 0; l < NL; l++) {
    const float* n1 = LP(l, 0);
    const float* n2 = LP(l, 1);
    const float* n3 = LP(l, 2);
    const float* bq = LP(l, 4);
    const float* bk_ = LP(l, 6);
    const float* bv = LP(l, 8);
    const float* bo = LP(l, 10);

    // ---- self attention ----
    rmsnorm_kernel<<<M, 128, 0, stream>>>(x, n1, xn, D);
    gemm(xn, wb[l][0], bq,  nullptr, qf, nullptr, M, D, D);
    gemm(xn, wb[l][1], bk_, nullptr, kf, nullptr, M, D, D);
    gemm(xn, wb[l][2], bv,  nullptr, nullptr, vb, M, D, D);
    rope_kernel<<<(npairs + 255) / 256, 256, 0, stream>>>(qf, kf, qb, kb, M, L, H, DH);
    attn_kernel<<<dim3(L / 64, B * H), 128, 0, stream>>>(qb, kb, vb, ob, nullptr, H, L, L, D, scale, 0);
    gemm(ob, wb[l][3], bo, x, x, nullptr, M, D, D);  // residual add

    // ---- cross attention (windowed by seg_ids) ----
    rmsnorm_kernel<<<M, 128, 0, stream>>>(x, n2, xn, D);
    gemm(xn,   wb[l][4], nullptr, nullptr, nullptr, qb,  M,  D, D);
    gemm(memb, wb[l][5], nullptr, nullptr, nullptr, kcb, Mh, D, D);
    gemm(memb, wb[l][6], nullptr, nullptr, nullptr, vcb, Mh, D, D);
    attn_kernel<<<dim3(L / 64, B * H), 128, 0, stream>>>(qb, kcb, vcb, ob, seg_ids, H, L, LH, D, scale, 128);
    gemm(ob, wb[l][7], nullptr, x, x, nullptr, M, D, D);

    // ---- SwiGLU FFN ----
    rmsnorm_kernel<<<M, 128, 0, stream>>>(x, n3, xn, D);
    gemm(xn, wb[l][8], nullptr, nullptr, h1, nullptr, M, DFF, D);
    gemm(xn, wb[l][9], nullptr, nullptr, h3, nullptr, M, DFF, D);
    swiglu_kernel<<<(unsigned)(((size_t)M * DFF + 255) / 256), 256, 0, stream>>>(h1, h3, hf, (size_t)M * DFF);
    gemm(hf, wb[l][10], nullptr, x, x, nullptr, M, D, DFF);
  }

  // ---- output head ----
  rmsnorm_kernel<<<M, 128, 0, stream>>>(x, fn_w, xn, D);
  gemm(xn, outw, out_b, nullptr, (float*)d_out, nullptr, M, KLO, D);
}
